// SharedMultiBarrelModel_50208167690753
// MI455X (gfx1250) — matrix operand order repaired and checked by reading, not compiled, not run on hardware
//
#include <hip/hip_runtime.h>
#include <hip/hip_fp16.h>

typedef __attribute__((ext_vector_type(16))) _Float16     v16h;
typedef __attribute__((ext_vector_type(8)))  float        v8f;
typedef __attribute__((ext_vector_type(4)))  unsigned int u32x4;

#define BATCH  64
#define NB     39      // barrels
#define MT     48      // padded M rows (3 x 16)
#define TSTEPS 64
#define CIN    64      // compressed channels
#define NPOP   8
#define PSZ    32
#define HID    256     // 8*32
#define NE     (NB*PSZ) // 1248 elements per pop per sample

// AlltoAll[j][i] connectivity mask (module constant in the reference)
__constant__ float ALLM[64] = {
 1,1,0,0,1,0,0,0,
 1,1,1,1,0,1,0,0,
 1,1,1,1,0,0,1,0,
 1,1,1,1,0,0,1,0,
 1,1,0,0,1,0,1,0,
 0,0,1,1,0,1,1,0,
 0,0,1,1,1,0,1,1,
 0,0,1,1,1,1,1,1 };

// ---------------------------------------------------------------------------
// WMMA fragment helpers (wave32, v_wmma_f32_16x16x32_f16; A/B use this code's contiguous k order)
// A (16x32 f16): lane L holds row M=L%16; half j -> K = (L/16)*16 + j
// B (32x16 f16): lane L holds col N=L%16; half j -> K = (L/16)*16 + j
// C/D (16x16 f32): lane L col N=L%16; vgpr r -> M = r + 8*(L/16)
// ---------------------------------------------------------------------------
__device__ __forceinline__ v16h loadA16(const _Float16* row, int kbase, int hi) {
  union { v16h v; u32x4 q[2]; } u;
  u.q[0] = *(const u32x4*)(row + kbase + hi * 16);       // halves 0..7
  u.q[1] = *(const u32x4*)(row + kbase + hi * 16 + 8);   // halves 8..15
  return u.v;
}
__device__ __forceinline__ v16h loadB16(const _Float16* g) {
  union { v16h v; u32x4 q[2]; } u;
  const u32x4* p = (const u32x4*)g;
  u.q[0] = p[0]; u.q[1] = p[1];
  return u.v;
}

// ---------------------------------------------------------------------------
// Kernel 0: precompute fused/masked weights as pre-swizzled f16 WMMA B-fragments
// Brec[i][ni][kb<10][lane][16] : rows 0..63 = th_w[i]+nagg_wx, rows 64..319 = masked Wm[:,i]
// Bn  [ni][kb<8][lane][16]     : nagg_ws.T (K=256) ; kb doubles as the per-pop delta block
// cbias[i][p], alpha[i][p], ro[i][p]
// ---------------------------------------------------------------------------
__global__ void prep_kernel(const float* th_w, const float* th_b, const float* pop_w,
                            const float* pop_b, const float* nagg_ws, const float* nagg_wx,
                            const float* nagg_b, const float* tau_adp, const float* tau_m,
                            _Float16* Brec, _Float16* Bn, float* cbias,
                            float* alpha, float* ro) {
  int idx = blockIdx.x * 256 + threadIdx.x;
  const int N_BREC = NPOP * 2 * 10 * 32 * 16;      // 81920
  const int N_BN   = 2 * 8 * 32 * 16;              // 8192
  if (idx < N_BREC) {
    int j = idx & 15, L = (idx >> 4) & 31;
    int rest = idx >> 9;
    int kb = rest % 10; rest /= 10;
    int ni = rest & 1;  int i = rest >> 1;
    int K = kb * 32 + ((L >> 4) & 1) * 16 + j;
    int p = ni * 16 + (L & 15);
    float v;
    if (K < 64) {
      v = th_w[(i * 32 + p) * 64 + K] + nagg_wx[p * 64 + K];
    } else {
      int q = K - 64; int j2 = q >> 5; q &= 31;
      v = pop_w[(((j2 * 8 + i) * 32 + p) * 32) + q] * ALLM[j2 * 8 + i];
    }
    Brec[idx] = (_Float16)v;
    return;
  }
  idx -= N_BREC;
  if (idx < N_BN) {
    int j = idx & 15, L = (idx >> 4) & 31;
    int rest = idx >> 9;
    int kb = rest & 7; // also = source pop for delta updates
    int K = kb * 32 + ((L >> 4) & 1) * 16 + j;
    int p = ((rest >> 3) & 1) * 16 + (L & 15);
    Bn[idx] = (_Float16)nagg_ws[p * 256 + K];
    return;
  }
  idx -= N_BN;
  if (idx < 256) {  // cbias
    int i = idx >> 5, p = idx & 31;
    float s = th_b[idx] + nagg_b[p];
    for (int j = 0; j < 8; ++j) s += pop_b[(j * 8 + i) * 32 + p] * ALLM[j * 8 + i];
    cbias[idx] = s;
    return;
  }
  idx -= 256;
  if (idx < 256) {
    alpha[idx] = expf(-1.0f / tau_m[idx]);
    ro[idx]    = expf(-1.0f / tau_adp[idx]);
  }
}

// ---------------------------------------------------------------------------
// Kernel 1: temporal compress conv1d(2->64, k=3, stride 2, pad 1), f32 -> f16
// x: [B][N][2][128]  ->  comp: [b][t][n][64]  (contiguous per (b,t) block)
// ---------------------------------------------------------------------------
__global__ void conv_kernel(const float* __restrict__ x, const float* __restrict__ w,
                            const float* __restrict__ bias, _Float16* __restrict__ comp) {
  int idx = blockIdx.x * 256 + threadIdx.x;
  const int TOT = BATCH * TSTEPS * NB * CIN;
  if (idx >= TOT) return;
  int o = idx & 63;
  int n = (idx >> 6) % NB;
  int t = (idx / (64 * NB)) % TSTEPS;
  int b = idx / (64 * NB * TSTEPS);
  float acc = bias[o];
  #pragma unroll
  for (int c = 0; c < 2; ++c) {
    #pragma unroll
    for (int k = 0; k < 3; ++k) {
      int ti = 2 * t + k - 1;
      if (ti >= 0 && ti < 128)
        acc += w[(o * 2 + c) * 3 + k] * x[((b * NB + n) * 2 + c) * 128 + ti];
    }
  }
  comp[((size_t)(b * TSTEPS + t) * NB + n) * 64 + o] = (_Float16)acc;
}

// ---------------------------------------------------------------------------
// Kernel 2: persistent scan. One block per batch sample, 8 wave32.
// Steady-state: per pop-step only 2 barriers; waves 0-5 run the K=320 WMMA GEMM
// for the NEXT pop while waves 6-7 apply the rank-32 asp update for the current
// pop (and hide the next timestep's global->LDS input copy).
// ---------------------------------------------------------------------------
__launch_bounds__(256, 1)
__global__ void scan_kernel(const _Float16* __restrict__ comp,
                            const _Float16* __restrict__ Brec,
                            const _Float16* __restrict__ Bn,
                            const float* __restrict__ cbias,
                            const float* __restrict__ alpha,
                            const float* __restrict__ ro,
                            const float* __restrict__ mem0,
                            const float* __restrict__ spike0,
                            const float* __restrict__ readout_w,
                            const float* __restrict__ readout_b,
                            float* __restrict__ out_soft,
                            float* __restrict__ h_out,
                            float* __restrict__ m_out) {
  __shared__ __align__(16) _Float16 inpA[2][MT][64];  // double-buffered inputs
  __shared__ __align__(16) _Float16 spA[MT][HID];     // all-pop spikes (A matrix)
  __shared__ __align__(16) _Float16 dSp[MT][32];      // spike delta of last pop
  __shared__ float curL[MT][32];                      // GEMM output tile buffer
  __shared__ float aspL[MT][32];                      // allsp @ nagg_ws.T (persistent)
  __shared__ float cbiasL[8][32], alphaL[8][32], roL[8][32];
  __shared__ float featL[128];
  __shared__ float logitsL[36];

  const int b    = blockIdx.x;
  const int tid  = threadIdx.x;
  const int lane = tid & 31;
  const int wv   = tid >> 5;
  const int hi   = (lane >> 4) & 1;

  // --- init LDS ---
  for (int i2 = tid; i2 < 2 * MT * 64; i2 += 256) ((_Float16*)inpA)[i2] = (_Float16)0.0f;
  for (int i2 = tid; i2 < MT * HID; i2 += 256)    ((_Float16*)spA)[i2]  = (_Float16)0.0f;
  for (int i2 = tid; i2 < MT * 32; i2 += 256)     ((_Float16*)dSp)[i2]  = (_Float16)0.0f;
  if (tid < 128) featL[tid] = 0.0f;
  cbiasL[tid >> 5][tid & 31] = cbias[tid];
  alphaL[tid >> 5][tid & 31] = alpha[tid];
  roL[tid >> 5][tid & 31]    = ro[tid];

  // --- per-thread register state (pop loop fully unrolled => static indexing) ---
  float memR[NPOP][5], bR[NPOP][5], accF[4][5];
  #pragma unroll
  for (int i = 0; i < 4; ++i)
    #pragma unroll
    for (int r = 0; r < 5; ++r) accF[i][r] = 0.0f;

  #pragma unroll
  for (int r = 0; r < 5; ++r) {
    int e = tid + 256 * r;
    if (e < NE) {
      int n = e >> 5, p = e & 31;
      #pragma unroll
      for (int i = 0; i < NPOP; ++i) {
        size_t si = ((size_t)(i * BATCH + b) * NB + n) * 32 + p;
        memR[i][r] = mem0[si];
        bR[i][r]   = 0.02f;                 // b0 = B_J0
        spA[n][i * 32 + p] = (_Float16)spike0[si];
      }
    }
  }
  __syncthreads();

  // --- prologue A: waves 0-5 build asp = allsp0 @ nagg_ws.T ; waves 6-7 stage t=0 ---
  if (wv < 6) {
    int mi = wv >> 1, ni = wv & 1;
    v8f c0 = {}, c1 = {};
    #pragma unroll
    for (int kb = 0; kb < 8; ++kb) {
      v16h a  = loadA16(&spA[mi * 16 + (lane & 15)][0], kb * 32, hi);
      v16h bf = loadB16(Bn + (((ni * 8 + kb) * 32 + lane) << 4));
      if (kb & 1)
        c1 = __builtin_amdgcn_wmma_f32_16x16x32_f16(false, a, false, bf, (short)0, c1, false, false);
      else
        c0 = __builtin_amdgcn_wmma_f32_16x16x32_f16(false, a, false, bf, (short)0, c0, false, false);
    }
    int col = ni * 16 + (lane & 15), mb = mi * 16 + hi * 8;
    #pragma unroll
    for (int r = 0; r < 8; ++r) aspL[mb + r][col] = c0[r] + c1[r];
  } else {
    // copy t=0 inputs into buffer 0 (read-once: nontemporal)
    const unsigned int* s32 = (const unsigned int*)(comp + (size_t)(b * TSTEPS) * NB * 64);
    unsigned int* d32 = (unsigned int*)&inpA[0][0][0];
    for (int i2 = tid - 192; i2 < NB * 64 / 2; i2 += 64)
      d32[i2] = __builtin_nontemporal_load(&s32[i2]);
  }
  __syncthreads();

  // --- prologue B: waves 0-5 run GEMM for (t=0, pop=0) ---
  if (wv < 6) {
    const int mi = wv >> 1, ni = wv & 1;
    const int m = mi * 16 + (lane & 15);
    const _Float16* bbase = Brec + (((size_t)(ni * 10) * 32 + lane) << 4);
    v8f c0 = {}, c1 = {};
    #pragma unroll
    for (int kb = 0; kb < 10; ++kb) {
      v16h a = (kb < 2) ? loadA16(&inpA[0][m][0], kb * 32, hi)
                        : loadA16(&spA[m][0], (kb - 2) * 32, hi);
      v16h bf = loadB16(bbase + ((kb * 32) << 4));
      if (kb & 1)
        c1 = __builtin_amdgcn_wmma_f32_16x16x32_f16(false, a, false, bf, (short)0, c1, false, false);
      else
        c0 = __builtin_amdgcn_wmma_f32_16x16x32_f16(false, a, false, bf, (short)0, c0, false, false);
    }
    int col = ni * 16 + (lane & 15), mb = mi * 16 + hi * 8;
    #pragma unroll
    for (int r = 0; r < 8; ++r) curL[mb + r][col] = c0[r] + c1[r];
  }
  __syncthreads();

  // =================== time scan: 2 barriers per pop-step ===================
  for (int t = 0; t < TSTEPS; ++t) {
    #pragma unroll
    for (int i = 0; i < NPOP; ++i) {
      // ---- phase A: LIF update + neighbor term + streamed outputs ----
      #pragma unroll
      for (int r = 0; r < 5; ++r) {
        int e = tid + 256 * r;
        if (e < NE) {
          int n = e >> 5, p = e & 31;
          int nm2 = (n >= 2) ? n - 2 : n + 37;
          int nm1 = (n >= 1) ? n - 1 : n + 38;
          int np1 = (n < 38) ? n + 1 : n - 38;
          int np2 = (n < 37) ? n + 2 : n - 37;
          float curv = curL[n][p] + cbiasL[i][p]
                     + 0.25f * (aspL[nm2][p] + aspL[nm1][p] + aspL[np1][p] + aspL[np2][p]);
          float spold = (float)spA[n][i * 32 + p];
          float roi = roL[i][p], al = alphaL[i][p];
          float bi  = roi * bR[i][r] + (1.0f - roi) * spold;
          float Bth = 0.02f + 1.8f * bi;
          float mem = memR[i][r] * al + (1.0f - al) * curv - Bth * spold;
          float sp  = (mem - Bth) > 0.0f ? 1.0f : 0.0f;
          memR[i][r] = mem;  bR[i][r] = bi;
          spA[n][i * 32 + p] = (_Float16)sp;
          dSp[n][p] = (_Float16)(sp - spold);
          size_t off = ((size_t)(b * TSTEPS + t) * NB + n) * HID + i * 32 + p;
          __builtin_nontemporal_store(sp, &h_out[off]);
          __builtin_nontemporal_store(mem + sp * Bth, &m_out[off]);
          if (i >= 4) accF[i - 4][r] += sp;
        }
      }
      __syncthreads();

      // ---- phase B (fused, overlapped across waves) ----
      if (wv < 6) {
        // next pop's fused input+recurrent GEMM [39,320] x [320,32]
        if (!(t == TSTEPS - 1 && i == NPOP - 1)) {
          const int inext = (i + 1) & 7;
          const int buf   = (i == NPOP - 1) ? ((t + 1) & 1) : (t & 1);
          const int mi = wv >> 1, ni = wv & 1;
          const int m = mi * 16 + (lane & 15);
          const _Float16* bbase = Brec + (((size_t)((inext * 2 + ni) * 10) * 32 + lane) << 4);
          v8f c0 = {}, c1 = {};
          #pragma unroll
          for (int kb = 0; kb < 10; ++kb) {
            v16h a = (kb < 2) ? loadA16(&inpA[buf][m][0], kb * 32, hi)
                              : loadA16(&spA[m][0], (kb - 2) * 32, hi);
            v16h bf = loadB16(bbase + ((kb * 32) << 4));
            if (kb & 1)
              c1 = __builtin_amdgcn_wmma_f32_16x16x32_f16(false, a, false, bf, (short)0, c1, false, false);
            else
              c0 = __builtin_amdgcn_wmma_f32_16x16x32_f16(false, a, false, bf, (short)0, c0, false, false);
          }
          int col = ni * 16 + (lane & 15), mb = mi * 16 + hi * 8;
          #pragma unroll
          for (int r = 0; r < 8; ++r) curL[mb + r][col] = c0[r] + c1[r];
        }
      } else {
        // rank-32 incremental update: asp += dSp @ nagg_ws[:, i*32:].T  (3 tiles/wave)
        int base = (wv - 6) * 3;
        #pragma unroll
        for (int q = 0; q < 3; ++q) {
          int tt = base + q;                 // 0..5
          int mi = tt >> 1, ni = tt & 1;
          int col = ni * 16 + (lane & 15), mb = mi * 16 + hi * 8;
          v8f c;
          #pragma unroll
          for (int r = 0; r < 8; ++r) c[r] = aspL[mb + r][col];
          v16h a  = loadA16(&dSp[mi * 16 + (lane & 15)][0], 0, hi);
          v16h bf = loadB16(Bn + (((ni * 8 + i) * 32 + lane) << 4));
          c = __builtin_amdgcn_wmma_f32_16x16x32_f16(false, a, false, bf, (short)0, c, false, false);
          #pragma unroll
          for (int r = 0; r < 8; ++r) aspL[mb + r][col] = c[r];
        }
        // hide next timestep's input staging + prefetch one further ahead
        if (i == 0 && t + 1 < TSTEPS) {
          const unsigned int* s32 =
              (const unsigned int*)(comp + (size_t)(b * TSTEPS + t + 1) * NB * 64);
          unsigned int* d32 = (unsigned int*)&inpA[(t + 1) & 1][0][0];
          for (int i2 = tid - 192; i2 < NB * 64 / 2; i2 += 64)
            d32[i2] = __builtin_nontemporal_load(&s32[i2]);
          if (t + 2 < TSTEPS && (tid - 192) < NB)
            __builtin_prefetch(comp + (size_t)(b * TSTEPS + t + 2) * NB * 64 + (tid - 192) * 64, 0, 0);
        }
      }
      __syncthreads();
    }
  }

  // =================== readout: pool L56 over (t,n), linear, softmax ===================
  #pragma unroll
  for (int i = 0; i < 4; ++i)
    #pragma unroll
    for (int r = 0; r < 5; ++r) {
      int e = tid + 256 * r;
      if (e < NE) atomicAdd(&featL[i * 32 + (e & 31)], accF[i][r]);
    }
  __syncthreads();
  if (tid < 36) {
    const float inv = 1.0f / (float)(TSTEPS * NB);
    float s = readout_b[tid];
    for (int k = 0; k < 128; ++k) s += featL[k] * inv * readout_w[tid * 128 + k];
    logitsL[tid] = s;
  }
  __syncthreads();
  if (tid < 36) {
    float mx = -3.4e38f;
    for (int k = 0; k < 36; ++k) mx = fmaxf(mx, logitsL[k]);
    float den = 0.0f;
    for (int k = 0; k < 36; ++k) den += __expf(logitsL[k] - mx);
    out_soft[b * 36 + tid] = __expf(logitsL[tid] - mx) / den;
  }
}

// ---------------------------------------------------------------------------
extern "C" void kernel_launch(void* const* d_in, const int* in_sizes, int n_in,
                              void* d_out, int out_size, void* d_ws, size_t ws_size,
                              hipStream_t stream) {
  const float* inputs    = (const float*)d_in[0];
  const float* conv_w    = (const float*)d_in[1];
  const float* conv_b    = (const float*)d_in[2];
  const float* th_w      = (const float*)d_in[3];
  const float* th_b      = (const float*)d_in[4];
  const float* pop_w     = (const float*)d_in[5];
  const float* pop_b     = (const float*)d_in[6];
  const float* nagg_ws   = (const float*)d_in[7];
  const float* nagg_wx   = (const float*)d_in[8];
  const float* nagg_b    = (const float*)d_in[9];
  const float* readout_w = (const float*)d_in[10];
  const float* readout_b = (const float*)d_in[11];
  const float* tau_adp   = (const float*)d_in[12];
  const float* tau_m     = (const float*)d_in[13];
  const float* mem0      = (const float*)d_in[14];
  const float* spike0    = (const float*)d_in[15];

  // workspace layout
  const size_t N_COMP = (size_t)BATCH * TSTEPS * NB * CIN;  // 10,223,616 halves
  const size_t N_BREC = NPOP * 2 * 10 * 32 * 16;            // 81,920
  const size_t N_BN   = 2 * 8 * 32 * 16;                    // 8,192
  _Float16* ws_comp = (_Float16*)d_ws;
  _Float16* ws_brec = ws_comp + N_COMP;
  _Float16* ws_bn   = ws_brec + N_BREC;
  float*    ws_cb   = (float*)(ws_bn + N_BN);
  float*    ws_al   = ws_cb + 256;
  float*    ws_ro   = ws_al + 256;

  float* outp  = (float*)d_out;
  float* h_out = outp + (size_t)BATCH * 36;
  float* m_out = h_out + (size_t)BATCH * TSTEPS * NB * HID;

  // 0) weight prep (fragments + fused biases + decay factors)
  {
    int total = (int)(N_BREC + N_BN + 256 + 256);
    prep_kernel<<<(total + 255) / 256, 256, 0, stream>>>(
        th_w, th_b, pop_w, pop_b, nagg_ws, nagg_wx, nagg_b, tau_adp, tau_m,
        ws_brec, ws_bn, ws_cb, ws_al, ws_ro);
  }
  // 1) temporal compress conv -> f16
  {
    int total = BATCH * TSTEPS * NB * CIN;
    conv_kernel<<<(total + 255) / 256, 256, 0, stream>>>(inputs, conv_w, conv_b, ws_comp);
  }
  // 2) persistent scan: one block per batch sample
  scan_kernel<<<BATCH, 256, 0, stream>>>(ws_comp, ws_brec, ws_bn, ws_cb, ws_al, ws_ro,
                                         mem0, spike0, readout_w, readout_b,
                                         outp, h_out, m_out);
}
